// PointTriNet_38517266710618
// MI455X (gfx1250) — compile-verified, matrix operand order checked by reading, not run on hardware
//
#include <hip/hip_runtime.h>
#include <hip/hip_bf16.h>
#include <math.h>

// ---------------------------------------------------------------------------
// Problem dimensions (from setup_inputs): B=2 V=10000 T=8192 Q=1024 K=32 K_T=32 O=4
// ---------------------------------------------------------------------------
constexpr int B_  = 2;
constexpr int V_  = 10000;
constexpr int T_  = 8192;
constexpr int Q_  = 1024;
constexpr int K_  = 32;
constexpr int KT_ = 32;
constexpr int O_  = 4;
constexpr int BQ  = B_ * Q_;       // 2048
constexpr int NP  = BQ * K_;       // 65536 rows for point-wise MLPs

// ---------------------------------------------------------------------------
// bf16 / WMMA types
// ---------------------------------------------------------------------------
typedef __bf16 bf16_t;
typedef __attribute__((ext_vector_type(16))) __bf16 v16bf;
typedef __attribute__((ext_vector_type(8)))  __bf16 v8bf;
typedef __attribute__((ext_vector_type(8)))  float  v8f;

// ---------------------------------------------------------------------------
// CDNA5 async global->LDS DMA (ASYNCcnt-tracked, no VGPR round-trip)
// ---------------------------------------------------------------------------
__device__ __forceinline__ void async_copy_b128(unsigned lds_addr, const void* gaddr) {
  asm volatile("global_load_async_to_lds_b128 %0, %1, off"
               :: "v"(lds_addr), "v"((unsigned long long)(size_t)gaddr)
               : "memory");
}
__device__ __forceinline__ void wait_async0() {
  asm volatile("s_wait_asynccnt 0" ::: "memory");
}

// ---------------------------------------------------------------------------
// small float3 helpers
// ---------------------------------------------------------------------------
struct F3 { float x, y, z; };
__device__ __forceinline__ F3 f3(float x, float y, float z) { return {x, y, z}; }
__device__ __forceinline__ F3 add3(F3 a, F3 b) { return {a.x + b.x, a.y + b.y, a.z + b.z}; }
__device__ __forceinline__ F3 sub3(F3 a, F3 b) { return {a.x - b.x, a.y - b.y, a.z - b.z}; }
__device__ __forceinline__ F3 scl3(F3 a, float s) { return {a.x * s, a.y * s, a.z * s}; }
__device__ __forceinline__ float dot3(F3 a, F3 b) { return a.x * b.x + a.y * b.y + a.z * b.z; }
__device__ __forceinline__ F3 crs3(F3 a, F3 b) {
  return {a.y * b.z - a.z * b.y, a.z * b.x - a.x * b.z, a.x * b.y - a.y * b.x};
}
__device__ __forceinline__ float len3(F3 a) { return sqrtf(dot3(a, a)); }
__device__ __forceinline__ F3 nrm3(F3 a) { float l = len3(a); return scl3(a, 1.0f / l); }

// generate_coords for one point against one (normalized) triangle -> 6 values
__device__ __forceinline__ void pt_coords(F3 t0, F3 t1, F3 t2, F3 p, float out6[6]) {
  F3 e1 = sub3(t1, t0), e2 = sub3(t2, t0);
  F3 an = scl3(crs3(e1, e2), 0.5f);
  float area = len3(an) + 1e-6f;
  F3 n = scl3(an, 1.0f / area);
  F3 bary = scl3(add3(add3(t0, t1), t2), 1.0f / 3.0f);
  F3 cen = sub3(p, bary);
  float nc = dot3(n, cen);
  F3 planar = sub3(p, scl3(n, nc));
  F3 a0 = sub3(t1, planar), b0 = sub3(t2, planar);
  F3 a1 = sub3(t2, planar), b1 = sub3(t0, planar);
  F3 a2 = sub3(t0, planar), b2 = sub3(t1, planar);
  float inva = 1.0f / area;
  float u = (0.5f * dot3(n, crs3(a0, b0)) + (1e-6f / 3.0f)) * inva;
  float v = (0.5f * dot3(n, crs3(a1, b1)) + (1e-6f / 3.0f)) * inva;
  float w = (0.5f * dot3(n, crs3(a2, b2)) + (1e-6f / 3.0f)) * inva;
  u = fminf(fmaxf(u, -5.0f), 5.0f);
  v = fminf(fmaxf(v, -5.0f), 5.0f);
  w = fminf(fmaxf(w, -5.0f), 5.0f);
  F3 bx = nrm3(e1);
  F3 by = nrm3(crs3(n, bx));
  out6[0] = dot3(bx, cen);
  out6[1] = dot3(by, cen);
  out6[2] = nc;
  out6[3] = u;
  out6[4] = v;
  out6[5] = w;
}

__device__ __forceinline__ void atomicMaxF(float* addr, float v) {
  unsigned int* ua = (unsigned int*)addr;
  unsigned int old = __hip_atomic_load(ua, __ATOMIC_RELAXED, __HIP_MEMORY_SCOPE_AGENT);
  while (true) {
    float f = __uint_as_float(old);
    if (f >= v) break;
    unsigned int assumed = old;
    old = atomicCAS(ua, assumed, __float_as_uint(v));
    if (old == assumed) break;
  }
}

__device__ __forceinline__ float hash_uniform(unsigned x) {
  x ^= x >> 17; x *= 0xed5ad4bbu;
  x ^= x >> 11; x *= 0xac4c1b51u;
  x ^= x >> 15; x *= 0x31848babu;
  x ^= x >> 14;
  return ((x >> 8) + 0.5f) * (1.0f / 16777216.0f);
}

// ---------------------------------------------------------------------------
// Kernel: fill f32 buffer
// ---------------------------------------------------------------------------
__global__ void fill_f32_k(float* p, float v, int n) {
  int i = blockIdx.x * blockDim.x + threadIdx.x;
  if (i < n) p[i] = v;
}

// ---------------------------------------------------------------------------
// Kernel: pack fp32 weight [Ksrc, N] -> bf16 WMMA-B layout [N/16][Kp/32][32][16]
// lane l, element e maps to (col = nt*16 + l%16, k = kt*32 + interleave(l/16, e))
// matching the 16-bit operand striping (K 0..7 | 8..15 | 16..23 | 24..31).
// ---------------------------------------------------------------------------
__global__ void pack_weight_k(const float* __restrict__ W, bf16_t* __restrict__ Wp,
                              int Ksrc, int Kp, int N) {
  size_t idx = (size_t)blockIdx.x * blockDim.x + threadIdx.x;
  size_t total = (size_t)(N >> 4) * (Kp >> 5) * 512;
  if (idx >= total) return;
  int e    = (int)(idx & 15);
  int lane = (int)((idx >> 4) & 31);
  size_t rest = idx >> 9;
  int kt = (int)(rest % (size_t)(Kp >> 5));
  int nt = (int)(rest / (size_t)(Kp >> 5));
  int col  = nt * 16 + (lane & 15);
  int half = lane >> 4;
  int k = kt * 32 + ((e < 8) ? (half * 8 + e) : (16 + half * 8 + (e - 8)));
  float v = (k < Ksrc) ? W[(size_t)k * N + col] : 0.0f;
  Wp[idx] = (bf16_t)v;
}

// ---------------------------------------------------------------------------
// Kernel: encode neighbor points vs (optionally rolled) query triangles.
// One thread per (b,q,k). Writes bf16 row [6 coords | 26 zero pad].
// ---------------------------------------------------------------------------
__global__ void encode_points_k(const float* __restrict__ qt,    // [B,Q,3,3]
                                const float* __restrict__ verts, // [B,V,3]
                                const int* __restrict__ pn_ind,  // [B,Q,K]
                                int shift,
                                bf16_t* __restrict__ coords)     // [NP,32]
{
  int idx = blockIdx.x * blockDim.x + threadIdx.x;
  if (idx >= NP) return;
  int bq = idx >> 5;          // K_ == 32
  int b  = bq >> 10;          // Q_ == 1024
  const float* tq = qt + (size_t)bq * 9;
  F3 tr[3];
#pragma unroll
  for (int c = 0; c < 3; ++c) {
    int s0 = (c + 3 - shift) % 3;  // jnp.roll(+1) applied `shift` times
    tr[c] = f3(tq[s0 * 3 + 0], tq[s0 * 3 + 1], tq[s0 * 3 + 2]);
  }
  F3 ctr = scl3(add3(add3(tr[0], tr[1]), tr[2]), 1.0f / 3.0f);
  float s = (len3(sub3(tr[0], ctr)) + len3(sub3(tr[1], ctr)) + len3(sub3(tr[2], ctr))) *
                (1.0f / 3.0f) + 1e-5f;
  int vi = pn_ind[idx];
  const float* vp = verts + ((size_t)b * V_ + vi) * 3;
  F3 p = f3(vp[0], vp[1], vp[2]);
  float inv = 1.0f / s;
  float c6[6];
  pt_coords(scl3(tr[0], inv), scl3(tr[1], inv), scl3(tr[2], inv), scl3(p, inv), c6);
  bf16_t* row = coords + (size_t)idx * 32;
#pragma unroll
  for (int c = 0; c < 6; ++c) row[c] = (bf16_t)c6[c];
#pragma unroll
  for (int c = 6; c < 32; ++c) row[c] = (bf16_t)0.0f;
}

// ---------------------------------------------------------------------------
// Kernel: encode neighbor triangles -> 13 features (min6|max6|prob), bf16 padded
// ---------------------------------------------------------------------------
__global__ void encode_tris_k(const float* __restrict__ qt,     // [B,Q,3,3]
                              const float* __restrict__ tpos,   // [B,T,3,3]
                              const float* __restrict__ tprob,  // [B,T]
                              const int* __restrict__ fn_ind,   // [B,Q,KT]
                              bf16_t* __restrict__ coords)      // [BQ*KT,32]
{
  int idx = blockIdx.x * blockDim.x + threadIdx.x;
  if (idx >= BQ * KT_) return;
  int bq = idx >> 5;
  int b  = bq >> 10;
  const float* tq = qt + (size_t)bq * 9;
  F3 tr[3];
#pragma unroll
  for (int c = 0; c < 3; ++c) tr[c] = f3(tq[c * 3 + 0], tq[c * 3 + 1], tq[c * 3 + 2]);
  F3 ctr = scl3(add3(add3(tr[0], tr[1]), tr[2]), 1.0f / 3.0f);
  float s = (len3(sub3(tr[0], ctr)) + len3(sub3(tr[1], ctr)) + len3(sub3(tr[2], ctr))) *
                (1.0f / 3.0f) + 1e-5f;
  float inv = 1.0f / s;
  F3 t0 = scl3(tr[0], inv), t1 = scl3(tr[1], inv), t2 = scl3(tr[2], inv);
  int f = fn_ind[idx];
  const float* fp = tpos + ((size_t)b * T_ + f) * 9;
  float mn[6], mx[6];
#pragma unroll
  for (int c = 0; c < 6; ++c) { mn[c] = 3.0e38f; mx[c] = -3.0e38f; }
#pragma unroll
  for (int j = 0; j < 3; ++j) {
    F3 p = scl3(f3(fp[j * 3 + 0], fp[j * 3 + 1], fp[j * 3 + 2]), inv);
    float c6[6];
    pt_coords(t0, t1, t2, p, c6);
#pragma unroll
    for (int c = 0; c < 6; ++c) { mn[c] = fminf(mn[c], c6[c]); mx[c] = fmaxf(mx[c], c6[c]); }
  }
  bf16_t* row = coords + (size_t)idx * 32;
#pragma unroll
  for (int c = 0; c < 6; ++c) row[c] = (bf16_t)mn[c];
#pragma unroll
  for (int c = 0; c < 6; ++c) row[6 + c] = (bf16_t)mx[c];
  row[12] = (bf16_t)tprob[(size_t)b * T_ + f];
#pragma unroll
  for (int c = 13; c < 32; ++c) row[c] = (bf16_t)0.0f;
}

// ---------------------------------------------------------------------------
// Kernel: WMMA GEMM, 16x64 output per wave, async-DMA double-buffered weights.
//   C[M,N] = act(A[M,Kp] @ W + bias)
//   Block = 8 waves = 8 consecutive M-tiles sharing ONE 64-wide N-strip.
//   The 4 B tiles (4 KB) of each k-step are staged global->LDS with
//   GLOBAL_LOAD_ASYNC_TO_LDS_B128 (ASYNCcnt), double-buffered so the DMA of
//   step kt+1 overlaps the 4 WMMAs of step kt; one barrier per k-step.
//   mode 0: store bf16 [M,N] (optional ReLU)
//   mode 1: fused max-pool over 32-row groups -> outMax[M/32, N]
//           (lane-local 8-row max + shfl_xor(16) -> one atomic per tile-col)
// Requires: (M/16) % 8 == 0, N % 64 == 0 (true for every layer here).
// ---------------------------------------------------------------------------
__global__ __launch_bounds__(256) void wmma_gemm_k(
    const bf16_t* __restrict__ A, const bf16_t* __restrict__ Wp,
    const float* __restrict__ bias, bf16_t* __restrict__ outB,
    float* __restrict__ outMax, int M, int Kp, int N, int relu, int mode) {
  __shared__ alignas(64) bf16_t sB[2][4 * 512];   // 2 x 4KB B strips (double buffer)

  int tid    = threadIdx.x;
  int waveId = tid >> 5;
  int lane   = tid & 31;
  int tilesM  = M >> 4;
  int blocksM = tilesM >> 3;
  int bm   = blockIdx.x % blocksM;
  int tn64 = blockIdx.x / blocksM;
  int tm   = (bm << 3) + waveId;

  int l15  = lane & 15;
  int half = lane >> 4;
  int nkt  = Kp >> 5;
  const bf16_t* a_row = A + (size_t)((tm << 4) + l15) * Kp;
  // cooperative staging: thread tid DMAs 8 bf16 (16 B) of the 4-tile strip
  int linear = tid * 8;
  int jmine  = linear >> 9;
  int wmine  = linear & 511;
  const bf16_t* b_src = Wp + ((size_t)(tn64 * 4 + jmine) * (size_t)nkt) * 512 + wmine;
  unsigned lds0 = (unsigned)(size_t)(&sB[0][linear]);
  unsigned lds1 = (unsigned)(size_t)(&sB[1][linear]);

  // prologue: DMA k-step 0 into buffer 0
  async_copy_b128(lds0, b_src);
  wait_async0();
  __syncthreads();

  v8f acc0 = {}, acc1 = {}, acc2 = {}, acc3 = {};
  for (int kt = 0; kt < nkt; ++kt) {
    // overlap: DMA next B strip into the other buffer while computing this one.
    // Safe: buffer (kt+1)&1 was last READ at step kt-1, and every wave passed
    // the kt-1 barrier before any wave issues this write.
    if (kt + 1 < nkt)
      async_copy_b128((kt & 1) ? lds0 : lds1, b_src + (size_t)(kt + 1) * 512);

    int k0 = kt << 5;
    v8bf alo = *(const v8bf*)(a_row + k0 + half * 8);
    v8bf ahi = *(const v8bf*)(a_row + k0 + 16 + half * 8);
    if (kt + 1 < nkt) __builtin_prefetch(a_row + k0 + 32, 0, 1);
    v16bf a;
#pragma unroll
    for (int e = 0; e < 8; ++e) { a[e] = alo[e]; a[8 + e] = ahi[e]; }

    const bf16_t* sb = sB[kt & 1];
    v16bf b0 = *(const v16bf*)(sb + 0 * 512 + lane * 16);
    v16bf b1 = *(const v16bf*)(sb + 1 * 512 + lane * 16);
    v16bf b2 = *(const v16bf*)(sb + 2 * 512 + lane * 16);
    v16bf b3 = *(const v16bf*)(sb + 3 * 512 + lane * 16);
    acc0 = __builtin_amdgcn_wmma_f32_16x16x32_bf16(false, a, false, b0, (short)0, acc0, false, false);
    acc1 = __builtin_amdgcn_wmma_f32_16x16x32_bf16(false, a, false, b1, (short)0, acc1, false, false);
    acc2 = __builtin_amdgcn_wmma_f32_16x16x32_bf16(false, a, false, b2, (short)0, acc2, false, false);
    acc3 = __builtin_amdgcn_wmma_f32_16x16x32_bf16(false, a, false, b3, (short)0, acc3, false, false);

    wait_async0();       // own DMA for kt+1 landed
    __syncthreads();     // everyone's DMA landed; reads of sB[kt&1] done
  }

  int baseRow = tm << 4;
  v8f accs[4] = {acc0, acc1, acc2, acc3};
#pragma unroll
  for (int j = 0; j < 4; ++j) {
    int col = (tn64 * 4 + j) * 16 + l15;
    float bv = bias ? bias[col] : 0.0f;
    v8f acc = accs[j];
    if (mode == 0) {
#pragma unroll
      for (int r = 0; r < 8; ++r) {
        float v = acc[r] + bv;
        if (relu) v = fmaxf(v, 0.0f);
        outB[(size_t)(baseRow + r + (half << 3)) * N + col] = (bf16_t)v;
      }
    } else {
      float m = -3.4e38f;
#pragma unroll
      for (int r = 0; r < 8; ++r) m = fmaxf(m, acc[r] + bv);
      float o = __shfl_xor(m, 16, 32);     // combine rows 0-7 with rows 8-15
      m = fmaxf(m, o);
      if (half == 0) atomicMaxF(&outMax[(size_t)(baseRow >> 5) * N + col], m);
    }
  }
}

// ---------------------------------------------------------------------------
// Kernel: build gc input: [BQ, 2048] bf16 = concat(pfm, tfm)
// ---------------------------------------------------------------------------
__global__ void gcat_k(const float* __restrict__ pfm, const float* __restrict__ tfm,
                       bf16_t* __restrict__ out) {
  size_t idx = (size_t)blockIdx.x * blockDim.x + threadIdx.x;
  if (idx >= (size_t)BQ * 2048) return;
  int r = (int)(idx >> 11);
  int c = (int)(idx & 2047);
  float v = (c < 1024) ? pfm[(size_t)r * 1024 + c] : tfm[(size_t)r * 1024 + (c - 1024)];
  out[idx] = (bf16_t)v;
}

// ---------------------------------------------------------------------------
// Kernel: gc head: sigmoid(h2 @ W3 + b3) with eps mix -> output_probs + d_out[0..BQ)
// ---------------------------------------------------------------------------
__global__ void gc_final_k(const bf16_t* __restrict__ h, const float* __restrict__ W,
                           const float* __restrict__ bias, float* __restrict__ outprobs,
                           float* __restrict__ dout) {
  int r = blockIdx.x * blockDim.x + threadIdx.x;
  if (r >= BQ) return;
  float acc = bias[0];
  const bf16_t* hr = h + (size_t)r * 256;
  for (int j = 0; j < 256; ++j) acc += (float)hr[j] * W[j];
  float sg = 1.0f / (1.0f + expf(-acc));
  float op = (1.0f - 1e-4f) * sg + 1e-4f * 0.5f;
  outprobs[r] = op;
  dout[r] = op;
}

// ---------------------------------------------------------------------------
// Kernel: build sel_in bf16 [NP,160] = [coords(6) | broadcast pfm_s(128) | pad]
// ---------------------------------------------------------------------------
__global__ void sel_build_k(const bf16_t* __restrict__ coords, const float* __restrict__ pfm_s,
                            bf16_t* __restrict__ out) {
  size_t idx = (size_t)blockIdx.x * blockDim.x + threadIdx.x;
  if (idx >= (size_t)NP * 160) return;
  int r = (int)(idx / 160);
  int c = (int)(idx - (size_t)r * 160);
  bf16_t v;
  if (c < 6)        v = coords[(size_t)r * 32 + c];
  else if (c < 134) v = (bf16_t)pfm_s[(size_t)(r >> 5) * 128 + (c - 6)];
  else              v = (bf16_t)0.0f;
  out[idx] = v;
}

// ---------------------------------------------------------------------------
// Kernel: sg head: sigmoid(h @ W4 + b4) -> point_probs [NP]
// ---------------------------------------------------------------------------
__global__ void sg_final_k(const bf16_t* __restrict__ h, const float* __restrict__ W,
                           const float* __restrict__ bias, float* __restrict__ pp) {
  int r = blockIdx.x * blockDim.x + threadIdx.x;
  if (r >= NP) return;
  float acc = bias[0];
  const bf16_t* hr = h + (size_t)r * 64;
#pragma unroll
  for (int j = 0; j < 64; ++j) acc += (float)hr[j] * W[j];
  pp[r] = 1.0f / (1.0f + expf(-acc));
}

// ---------------------------------------------------------------------------
// Kernel: per-(b,q) masked Gumbel top-O selection -> gen tris/probs into d_out
// d_out layout: [BQ output_probs][BQ*12*3 tris-as-float][BQ*12 probs]
// ---------------------------------------------------------------------------
__global__ void select_k(const float* __restrict__ point_probs, const int* __restrict__ pn_ind,
                         const int* __restrict__ qti, const float* __restrict__ outprobs,
                         float* __restrict__ dout, int gen_i) {
  int bq = blockIdx.x * blockDim.x + threadIdx.x;
  if (bq >= BQ) return;
  int shift = gen_i;
  int ci[3];
#pragma unroll
  for (int j = 0; j < 3; ++j) ci[j] = qti[(size_t)bq * 3 + ((j + 3 - shift) % 3)];
  float pp[32], sc[32];
#pragma unroll
  for (int k = 0; k < 32; ++k) {
    pp[k] = point_probs[(size_t)bq * 32 + k];
    float sp = pp[k] + 1e-4f;
    int vi = pn_ind[(size_t)bq * 32 + k];
    if (vi == ci[0] || vi == ci[1] || vi == ci[2]) sp = 0.0f;
    float u = hash_uniform((unsigned)(gen_i * 0x9e3779b9u) ^ (unsigned)(bq * 32 + k) * 0x85ebca6bu ^ 0x6d2b79f5u);
    float g = -logf(-logf(u));
    sc[k] = (sp > 0.0f) ? (logf(fmaxf(sp, 1e-30f)) + g) : -3.4e38f;
  }
  float* gt = dout + BQ;
  float* gp = dout + BQ + (size_t)BQ * 12 * 3;
#pragma unroll
  for (int o = 0; o < O_; ++o) {
    int kb = 0;
    float best = sc[0];
#pragma unroll
    for (int k = 1; k < 32; ++k)
      if (sc[k] > best) { best = sc[k]; kb = k; }
    sc[kb] = -3.4e38f;
    int ni = pn_ind[(size_t)bq * 32 + kb];
    float np = ((1.0f - 1e-4f) * pp[kb] + 1e-4f * 0.5f) * outprobs[bq];
    size_t t = (size_t)bq * 12 + gen_i * 4 + o;
    gt[t * 3 + 0] = (float)ci[0];
    gt[t * 3 + 1] = (float)ci[1];
    gt[t * 3 + 2] = (float)ni;
    gp[t] = np;
  }
}

// ---------------------------------------------------------------------------
// Host orchestration
// ---------------------------------------------------------------------------
extern "C" void kernel_launch(void* const* d_in, const int* in_sizes, int n_in,
                              void* d_out, int out_size, void* d_ws, size_t ws_size,
                              hipStream_t stream) {
  (void)in_sizes; (void)n_in; (void)out_size; (void)ws_size;
  const float* verts  = (const float*)d_in[0];
  const float* tpos   = (const float*)d_in[1];
  const float* tprob  = (const float*)d_in[2];
  const float* qt     = (const float*)d_in[3];
  const int*   qti    = (const int*)d_in[4];
  const int*   pn_ind = (const int*)d_in[6];
  const int*   fn_ind = (const int*)d_in[7];
  // params flattened in dict insertion order: pc, tc, gc, ps, sg ; each (W,b) pair
  const float* pcW1 = (const float*)d_in[9];  const float* pcB1 = (const float*)d_in[10];
  const float* pcW2 = (const float*)d_in[11]; const float* pcB2 = (const float*)d_in[12];
  const float* pcW3 = (const float*)d_in[13]; const float* pcB3 = (const float*)d_in[14];
  const float* tcW1 = (const float*)d_in[15]; const float* tcB1 = (const float*)d_in[16];
  const float* tcW2 = (const float*)d_in[17]; const float* tcB2 = (const float*)d_in[18];
  const float* tcW3 = (const float*)d_in[19]; const float* tcB3 = (const float*)d_in[20];
  const float* gcW1 = (const float*)d_in[21]; const float* gcB1 = (const float*)d_in[22];
  const float* gcW2 = (const float*)d_in[23]; const float* gcB2 = (const float*)d_in[24];
  const float* gcW3 = (const float*)d_in[25]; const float* gcB3 = (const float*)d_in[26];
  const float* psW1 = (const float*)d_in[27]; const float* psB1 = (const float*)d_in[28];
  const float* psW2 = (const float*)d_in[29]; const float* psB2 = (const float*)d_in[30];
  const float* psW3 = (const float*)d_in[31]; const float* psB3 = (const float*)d_in[32];
  const float* sgW1 = (const float*)d_in[33]; const float* sgB1 = (const float*)d_in[34];
  const float* sgW2 = (const float*)d_in[35]; const float* sgB2 = (const float*)d_in[36];
  const float* sgW3 = (const float*)d_in[37]; const float* sgB3 = (const float*)d_in[38];
  const float* sgW4 = (const float*)d_in[39]; const float* sgB4 = (const float*)d_in[40];

  float* out = (float*)d_out;

  // --- carve workspace ---
  char* ws = (char*)d_ws;
  size_t off = 0;
  auto carve = [&](size_t bytes) -> char* {
    char* p = ws + off;
    off = (off + bytes + 255) & ~(size_t)255;
    return p;
  };
  bf16_t* wp_pc1 = (bf16_t*)carve((size_t)32 * 64 * 2);
  bf16_t* wp_pc2 = (bf16_t*)carve((size_t)64 * 128 * 2);
  bf16_t* wp_pc3 = (bf16_t*)carve((size_t)128 * 1024 * 2);
  bf16_t* wp_tc1 = (bf16_t*)carve((size_t)32 * 64 * 2);
  bf16_t* wp_tc2 = (bf16_t*)carve((size_t)64 * 128 * 2);
  bf16_t* wp_tc3 = (bf16_t*)carve((size_t)128 * 1024 * 2);
  bf16_t* wp_gc1 = (bf16_t*)carve((size_t)2048 * 512 * 2);
  bf16_t* wp_gc2 = (bf16_t*)carve((size_t)512 * 256 * 2);
  bf16_t* wp_ps1 = (bf16_t*)carve((size_t)32 * 64 * 2);
  bf16_t* wp_ps2 = (bf16_t*)carve((size_t)64 * 64 * 2);
  bf16_t* wp_ps3 = (bf16_t*)carve((size_t)64 * 128 * 2);
  bf16_t* wp_sg1 = (bf16_t*)carve((size_t)160 * 128 * 2);
  bf16_t* wp_sg2 = (bf16_t*)carve((size_t)128 * 64 * 2);
  bf16_t* wp_sg3 = (bf16_t*)carve((size_t)64 * 64 * 2);
  bf16_t* coords_p = (bf16_t*)carve((size_t)NP * 32 * 2);
  bf16_t* coords_t = (bf16_t*)carve((size_t)NP * 32 * 2);
  float*  pfm      = (float*)carve((size_t)BQ * 1024 * 4);
  float*  tfm      = (float*)carve((size_t)BQ * 1024 * 4);
  float*  pfm_s    = (float*)carve((size_t)BQ * 128 * 4);
  float*  outprobs = (float*)carve((size_t)BQ * 4);
  float*  ppoint   = (float*)carve((size_t)NP * 4);
  bf16_t* act0     = (bf16_t*)carve((size_t)NP * 160 * 2);
  bf16_t* act1     = (bf16_t*)carve((size_t)NP * 128 * 2);
  bf16_t* act2     = (bf16_t*)carve((size_t)NP * 128 * 2);

  auto pack = [&](const float* W, bf16_t* Wp, int Ks, int Kp, int N) {
    size_t tot = (size_t)(N / 16) * (Kp / 32) * 512;
    pack_weight_k<<<(unsigned)((tot + 255) / 256), 256, 0, stream>>>(W, Wp, Ks, Kp, N);
  };
  auto gemm = [&](const bf16_t* A, const bf16_t* Wp, const float* bias, bf16_t* oB,
                  float* oM, int M, int Kp, int N, int relu, int mode) {
    int grid = ((M / 16) / 8) * (N / 64);   // (M/16)%8==0 and N%64==0 for all layers
    wmma_gemm_k<<<grid, 256, 0, stream>>>(A, Wp, bias, oB, oM, M, Kp, N, relu, mode);
  };

  // --- weight packing (bf16, WMMA B-operand swizzle, K padded) ---
  pack(pcW1, wp_pc1, 6, 32, 64);     pack(pcW2, wp_pc2, 64, 64, 128);
  pack(pcW3, wp_pc3, 128, 128, 1024);
  pack(tcW1, wp_tc1, 13, 32, 64);    pack(tcW2, wp_tc2, 64, 64, 128);
  pack(tcW3, wp_tc3, 128, 128, 1024);
  pack(gcW1, wp_gc1, 2048, 2048, 512); pack(gcW2, wp_gc2, 512, 512, 256);
  pack(psW1, wp_ps1, 6, 32, 64);     pack(psW2, wp_ps2, 64, 64, 64);
  pack(psW3, wp_ps3, 64, 64, 128);
  pack(sgW1, wp_sg1, 134, 160, 128); pack(sgW2, wp_sg2, 128, 128, 64);
  pack(sgW3, wp_sg3, 64, 64, 64);

  // --- pooled-feature init ---
  fill_f32_k<<<(BQ * 1024 + 255) / 256, 256, 0, stream>>>(pfm, -3.0e38f, BQ * 1024);
  fill_f32_k<<<(BQ * 1024 + 255) / 256, 256, 0, stream>>>(tfm, -3.0e38f, BQ * 1024);

  // --- geometric encode ---
  encode_points_k<<<(NP + 255) / 256, 256, 0, stream>>>(qt, verts, pn_ind, 0, coords_p);
  encode_tris_k<<<(NP + 255) / 256, 256, 0, stream>>>(qt, tpos, tprob, fn_ind, coords_t);

  // --- pc MLP: 32(pad6)->64->128->1024, fused max over K into pfm ---
  gemm(coords_p, wp_pc1, pcB1, act1, nullptr, NP, 32, 64, 1, 0);
  gemm(act1, wp_pc2, pcB2, act2, nullptr, NP, 64, 128, 1, 0);
  gemm(act2, wp_pc3, pcB3, nullptr, pfm, NP, 128, 1024, 0, 1);

  // --- tc MLP: 32(pad13)->64->128->1024, fused max into tfm ---
  gemm(coords_t, wp_tc1, tcB1, act1, nullptr, NP, 32, 64, 1, 0);
  gemm(act1, wp_tc2, tcB2, act2, nullptr, NP, 64, 128, 1, 0);
  gemm(act2, wp_tc3, tcB3, nullptr, tfm, NP, 128, 1024, 0, 1);

  // --- gc MLP: concat(2048)->512->256->sigmoid head ---
  {
    size_t tot = (size_t)BQ * 2048;
    gcat_k<<<(unsigned)((tot + 255) / 256), 256, 0, stream>>>(pfm, tfm, act0);
  }
  gemm(act0, wp_gc1, gcB1, act1, nullptr, BQ, 2048, 512, 1, 0);
  gemm(act1, wp_gc2, gcB2, act2, nullptr, BQ, 512, 256, 1, 0);
  gc_final_k<<<(BQ + 255) / 256, 256, 0, stream>>>(act2, gcW3, gcB3, outprobs, out);

  // --- generation loop: 3 rolled sides ---
  for (int i = 0; i < 3; ++i) {
    if (i != 0)
      encode_points_k<<<(NP + 255) / 256, 256, 0, stream>>>(qt, verts, pn_ind, i, coords_p);
    fill_f32_k<<<(BQ * 128 + 255) / 256, 256, 0, stream>>>(pfm_s, -3.0e38f, BQ * 128);
    // ps MLP: 32(pad6)->64->64->128, fused max into pfm_s
    gemm(coords_p, wp_ps1, psB1, act1, nullptr, NP, 32, 64, 1, 0);
    gemm(act1, wp_ps2, psB2, act2, nullptr, NP, 64, 64, 1, 0);
    gemm(act2, wp_ps3, psB3, nullptr, pfm_s, NP, 64, 128, 0, 1);
    // sel_in = [coords | broadcast pooled | pad] -> sg MLP 160->128->64->64->sigmoid
    {
      size_t tot = (size_t)NP * 160;
      sel_build_k<<<(unsigned)((tot + 255) / 256), 256, 0, stream>>>(coords_p, pfm_s, act0);
    }
    gemm(act0, wp_sg1, sgB1, act1, nullptr, NP, 160, 128, 1, 0);
    gemm(act1, wp_sg2, sgB2, act2, nullptr, NP, 128, 64, 1, 0);
    gemm(act2, wp_sg3, sgB3, act0, nullptr, NP, 64, 64, 1, 0);
    sg_final_k<<<(NP + 255) / 256, 256, 0, stream>>>(act0, sgW4, sgB4, ppoint);
    // masked Gumbel top-4 -> outputs
    select_k<<<(BQ + 255) / 256, 256, 0, stream>>>(ppoint, pn_ind, qti, outprobs, out, i);
  }
}
